// TokenVisionTransformerMoE_44014824849490
// MI455X (gfx1250) — compile-verified, matrix operand order checked by reading, not run on hardware
//
#include <hip/hip_runtime.h>
#include <hip/hip_bf16.h>
#include <math.h>

typedef __attribute__((ext_vector_type(16))) _Float16 v16h;
typedef __attribute__((ext_vector_type(8)))  _Float16 v8h;
typedef __attribute__((ext_vector_type(4)))  _Float16 v4h;
typedef __attribute__((ext_vector_type(8)))  float    v8f;

#define BATCH   16
#define NTOK    197
#define ROWS    (BATCH*NTOK)   /* 3152 */
#define DMODEL  768
#define NH_     12
#define DHEAD   64
#define NLAYERS 4
#define NEXPERT 8
#define HID_    3072
#define NPATCH  196
#define GRID_   14
#define NPAD    224            /* keys padded to 7*32 for WMMA k-loop */
#define PROWS   208            /* 13 q-tiles * 16 */

// token (t % 40) -> expert id
__device__ __constant__ int GMAP[40] = {
  0,1,2,3,4,4,5,5,5,3,3,3,2,2,2,2,2,1,1,1,
  1,6,6,6,6,6,6,7,7,7,7,7,7,7,7,7,7,7,5,6 };
__device__ __constant__ int EOFF_D[8] = {0,5,30,60,80,90,109,143};
__device__ __constant__ int ECNT_D[8] = {5,25,30,20,10,19,34,54};

// ---------------------------------------------------------------- WMMA helpers
__device__ inline v16h frag_ld(const _Float16* row, int kh) {
  // CDNA5 16-bit A/B fragment: elements 0..7 -> K = kh*8+e, 8..15 -> 16+kh*8+e
  const v8h lo = *(const v8h*)(row + kh * 8);
  const v8h hi = *(const v8h*)(row + 16 + kh * 8);
  v16h f;
#pragma unroll
  for (int i = 0; i < 8; ++i) { f[i] = lo[i]; f[8 + i] = hi[i]; }
  return f;
}

__device__ inline v8f wmma32(v16h a, v16h b, v8f c) {
  return __builtin_amdgcn_wmma_f32_16x16x32_f16(false, a, false, b, (short)0, c,
                                                false, false);
}

__device__ inline void cvt_store4(_Float16* p, float4 v) {
  v4h h;
  h[0] = (_Float16)v.x; h[1] = (_Float16)v.y;
  h[2] = (_Float16)v.z; h[3] = (_Float16)v.w;
  *(v4h*)p = h;   // 8-byte ds_store_b64
}

__device__ inline float gelu_exact(float v) {
  return 0.5f * v * (1.0f + erff(v * 0.70710678118654752f));
}

// ------------------------------------------------ expert row map (flat gather)
// rmap[region_row] = b*197 + token ; region for expert e starts at 16*EOFF[e],
// rows within region ordered row = b*cnt + i  (matches reference y[:, idx]).
// Serial part only builds the 197-entry token list; the 3152-entry map fill is
// parallel.
__global__ __launch_bounds__(256) void build_maps_kernel(int* __restrict__ rmap) {
  __shared__ int emap[NTOK];
  if (threadIdx.x == 0) {
    int c[8];
#pragma unroll
    for (int e = 0; e < 8; ++e) c[e] = 0;
    for (int t = 0; t < NTOK; ++t) {
      int e = GMAP[t % 40];
      emap[EOFF_D[e] + c[e]] = t;
      c[e]++;
    }
  }
  __syncthreads();
  for (int j = threadIdx.x; j < ROWS; j += blockDim.x) {
    int e = 0;
    while (e < 7 && j >= 16 * EOFF_D[e + 1]) ++e;
    int cnt = ECNT_D[e];
    int loc = j - 16 * EOFF_D[e];
    rmap[j] = (loc / cnt) * NTOK + emap[EOFF_D[e] + loc % cnt];
  }
}

// ---------------------------------------------------------------- cls + pos row
__global__ void cls_pos_kernel(const float* __restrict__ cls,
                               const float* __restrict__ pos,
                               float* __restrict__ t) {
  int b = blockIdx.x;
  for (int d = threadIdx.x; d < DMODEL; d += blockDim.x)
    t[(size_t)b * NTOK * DMODEL + d] = cls[d] + pos[d];
}

// ---------------------------------------------------------------- LayerNorm
__global__ __launch_bounds__(256) void ln_kernel(const float* __restrict__ in,
                                                 float* __restrict__ out,
                                                 const float* __restrict__ g,
                                                 const float* __restrict__ b) {
  __shared__ float red[256];
  const int tid = threadIdx.x;
  const float* p = in + (size_t)blockIdx.x * DMODEL;
  float x0 = p[tid], x1 = p[tid + 256], x2 = p[tid + 512];
  red[tid] = x0 + x1 + x2;
  __syncthreads();
  for (int st = 128; st > 0; st >>= 1) {
    if (tid < st) red[tid] += red[tid + st];
    __syncthreads();
  }
  float mu = red[0] * (1.0f / DMODEL);
  __syncthreads();
  float d0 = x0 - mu, d1 = x1 - mu, d2 = x2 - mu;
  red[tid] = d0 * d0 + d1 * d1 + d2 * d2;
  __syncthreads();
  for (int st = 128; st > 0; st >>= 1) {
    if (tid < st) red[tid] += red[tid + st];
    __syncthreads();
  }
  float rstd = rsqrtf(red[0] * (1.0f / DMODEL) + 1e-6f);
  float* q = out + (size_t)blockIdx.x * DMODEL;
  q[tid]       = d0 * rstd * g[tid]       + b[tid];
  q[tid + 256] = d1 * rstd * g[tid + 256] + b[tid + 256];
  q[tid + 512] = d2 * rstd * g[tid + 512] + b[tid + 512];
}

// ---------------------------------------------------------------- generic GEMM
// 64x256 block tile, 8 waves as 2(M) x 4(N), wave tile 32x64 (2x4 WMMA tiles),
// K-tile 32, fp32->f16 convert while staging, fp32 accumulate. Double-buffered
// LDS: one barrier per k-tile; HBM fetch of k+1 overlaps WMMA of k, LDS store
// of k+1 follows the WMMAs.
// GATHER: A row = rmap[row]            (MoE token gather)
// LAYB  : 0 = W[n][k] (y @ W.T),  1 = W[k][n] (y @ W)
// EPI   : 1 = bias store, 2 = gelu(bias) store, 3 = bias + residual add,
//         4 = bias + scatter-add into residual via rmap
template <int GATHER, int LAYB, int EPI>
__global__ __launch_bounds__(256) void gemm_kernel(
    const float* __restrict__ A, const float* __restrict__ Bw,
    const float* __restrict__ bias, float* __restrict__ C, int M, int N, int K,
    const int* __restrict__ rmap) {
  __shared__ _Float16 As[2][64 * 40];
  __shared__ _Float16 Bs[2][256 * 40];
  const int tid = threadIdx.x;
  const int lane = tid & 31, wid = tid >> 5;
  const int wm = (wid >> 2) * 32, wn = (wid & 3) * 64;
  const int r = lane & 15, kh = lane >> 4;
  const int m0 = blockIdx.x * 64, n0 = blockIdx.y * 256;

  // A staging: 64 rows x 32 k, thread -> rows (tid>>3, +32), k chunk (tid&7)*4
  const int ar0 = tid >> 3, ak = (tid & 7) * 4;
  const float* aptr[2];
#pragma unroll
  for (int i = 0; i < 2; ++i) {
    int rg = m0 + ar0 + 32 * i;
    if (rg < M) {
      int src = GATHER ? rmap[rg] : rg;
      aptr[i] = A + (size_t)src * K + ak;
    } else {
      aptr[i] = nullptr;
    }
  }

  float4 aR[2], bR[8];

  auto fetch = [&](int kt) {
#pragma unroll
    for (int i = 0; i < 2; ++i)
      aR[i] = aptr[i] ? *(const float4*)(aptr[i] + kt * 32)
                      : make_float4(0.f, 0.f, 0.f, 0.f);
    if (LAYB == 0) {
      const int bn = tid >> 3, bk = (tid & 7) * 4;
#pragma unroll
      for (int i = 0; i < 8; ++i)
        bR[i] = *(const float4*)(Bw + (size_t)(n0 + bn + 32 * i) * K + kt * 32 + bk);
    } else {
      const int bn = (tid & 63) * 4, bk = tid >> 6;
#pragma unroll
      for (int i = 0; i < 8; ++i)
        bR[i] = *(const float4*)(Bw + (size_t)(kt * 32 + bk + 4 * i) * N + n0 + bn);
    }
  };

  auto stage = [&](int p) {
#pragma unroll
    for (int i = 0; i < 2; ++i)
      cvt_store4(&As[p][(ar0 + 32 * i) * 40 + ak], aR[i]);
    if (LAYB == 0) {
      const int bn = tid >> 3, bk = (tid & 7) * 4;
#pragma unroll
      for (int i = 0; i < 8; ++i)
        cvt_store4(&Bs[p][(bn + 32 * i) * 40 + bk], bR[i]);
    } else {
      const int bn = (tid & 63) * 4, bk = tid >> 6;
#pragma unroll
      for (int i = 0; i < 8; ++i) {
        int kk = bk + 4 * i;
        Bs[p][(bn + 0) * 40 + kk] = (_Float16)bR[i].x;
        Bs[p][(bn + 1) * 40 + kk] = (_Float16)bR[i].y;
        Bs[p][(bn + 2) * 40 + kk] = (_Float16)bR[i].z;
        Bs[p][(bn + 3) * 40 + kk] = (_Float16)bR[i].w;
      }
    }
  };

  v8f acc[2][4];
#pragma unroll
  for (int mi = 0; mi < 2; ++mi)
#pragma unroll
    for (int ni = 0; ni < 4; ++ni)
#pragma unroll
      for (int q = 0; q < 8; ++q) acc[mi][ni][q] = 0.0f;

  const int nkt = K >> 5;
  fetch(0);
  stage(0);
  for (int kt = 0; kt < nkt; ++kt) {
    __syncthreads();
    const int p = kt & 1;
    if (kt + 1 < nkt) fetch(kt + 1);            // HBM overlaps WMMA below
    v16h af[2], bf[4];
#pragma unroll
    for (int mi = 0; mi < 2; ++mi)
      af[mi] = frag_ld(&As[p][(wm + mi * 16 + r) * 40], kh);
#pragma unroll
    for (int ni = 0; ni < 4; ++ni)
      bf[ni] = frag_ld(&Bs[p][(wn + ni * 16 + r) * 40], kh);
#pragma unroll
    for (int mi = 0; mi < 2; ++mi)
#pragma unroll
      for (int ni = 0; ni < 4; ++ni)
        acc[mi][ni] = wmma32(af[mi], bf[ni], acc[mi][ni]);
    if (kt + 1 < nkt) stage(1 - p);             // fill other buffer
  }

#pragma unroll
  for (int mi = 0; mi < 2; ++mi)
#pragma unroll
    for (int ni = 0; ni < 4; ++ni)
#pragma unroll
      for (int v = 0; v < 8; ++v) {
        int rg = m0 + wm + mi * 16 + 8 * kh + v;
        if (rg >= M) continue;
        int cg = n0 + wn + ni * 16 + r;
        float val = acc[mi][ni][v] + bias[cg];
        if (EPI == 2) val = gelu_exact(val);
        if (EPI == 1 || EPI == 2) {
          C[(size_t)rg * N + cg] = val;
        } else if (EPI == 3) {
          C[(size_t)rg * N + cg] += val;
        } else if (EPI == 4) {
          int rr = rmap[rg];
          C[(size_t)rr * DMODEL + cg] += val;
        }
      }
}

// ---------------------------------------------------------------- patch embed
// t[b, 1+p, :] = patches[b,p,:] @ patch_w.T + patch_b + pos[1+p,:]
// 64x128 block, 8 waves 2x4, wave tile 32x32, double-buffered LDS.
__global__ __launch_bounds__(256) void patch_embed_kernel(
    const float* __restrict__ x, const float* __restrict__ Wp,
    const float* __restrict__ bp, const float* __restrict__ pos,
    float* __restrict__ t) {
  __shared__ _Float16 As[2][64 * 40];
  __shared__ _Float16 Bs[2][128 * 40];
  const int tid = threadIdx.x;
  const int lane = tid & 31, wid = tid >> 5;
  const int wm = (wid >> 2) * 32, wn = (wid & 3) * 32;
  const int r = lane & 15, kh = lane >> 4;
  const int m0 = blockIdx.x * 64, n0 = blockIdx.y * 128;

  const int ar0 = tid >> 3, ak = (tid & 7) * 4;
  int rb[2], ry[2], rx[2];
#pragma unroll
  for (int i = 0; i < 2; ++i) {
    int rg = m0 + ar0 + 32 * i;                 // < 3136 always
    int b = rg / NPATCH, p = rg % NPATCH;
    rb[i] = b; ry[i] = p / GRID_; rx[i] = p % GRID_;
  }

  float4 aR[2], bR[4];
  auto fetch = [&](int kt) {
    int kglob = kt * 32 + ak;
    int c = kglob >> 8, rem = kglob & 255, py = rem >> 4, px = rem & 15;
#pragma unroll
    for (int i = 0; i < 2; ++i)
      aR[i] = *(const float4*)(x + (((size_t)rb[i] * 3 + c) * 224 +
                                    ry[i] * 16 + py) * 224 + rx[i] * 16 + px);
    const int bn = tid >> 3;
#pragma unroll
    for (int i = 0; i < 4; ++i)
      bR[i] = *(const float4*)(Wp + (size_t)(n0 + bn + 32 * i) * DMODEL +
                               kt * 32 + ak);
  };
  auto stage = [&](int p) {
#pragma unroll
    for (int i = 0; i < 2; ++i)
      cvt_store4(&As[p][(ar0 + 32 * i) * 40 + ak], aR[i]);
    const int bn = tid >> 3;
#pragma unroll
    for (int i = 0; i < 4; ++i)
      cvt_store4(&Bs[p][(bn + 32 * i) * 40 + ak], bR[i]);
  };

  v8f acc[2][2];
#pragma unroll
  for (int mi = 0; mi < 2; ++mi)
#pragma unroll
    for (int ni = 0; ni < 2; ++ni)
#pragma unroll
      for (int q = 0; q < 8; ++q) acc[mi][ni][q] = 0.0f;

  fetch(0);
  stage(0);
  for (int kt = 0; kt < 24; ++kt) {
    __syncthreads();
    const int p = kt & 1;
    if (kt + 1 < 24) fetch(kt + 1);
    v16h af[2], bf[2];
#pragma unroll
    for (int mi = 0; mi < 2; ++mi)
      af[mi] = frag_ld(&As[p][(wm + mi * 16 + r) * 40], kh);
#pragma unroll
    for (int ni = 0; ni < 2; ++ni)
      bf[ni] = frag_ld(&Bs[p][(wn + ni * 16 + r) * 40], kh);
#pragma unroll
    for (int mi = 0; mi < 2; ++mi)
#pragma unroll
      for (int ni = 0; ni < 2; ++ni)
        acc[mi][ni] = wmma32(af[mi], bf[ni], acc[mi][ni]);
    if (kt + 1 < 24) stage(1 - p);
  }

#pragma unroll
  for (int mi = 0; mi < 2; ++mi)
#pragma unroll
    for (int ni = 0; ni < 2; ++ni)
#pragma unroll
      for (int v = 0; v < 8; ++v) {
        int rg = m0 + wm + mi * 16 + 8 * kh + v;
        int b = rg / NPATCH, p = rg % NPATCH;
        int cg = n0 + wn + ni * 16 + r;
        float val = acc[mi][ni][v] + bp[cg] + pos[(size_t)(1 + p) * DMODEL + cg];
        t[((size_t)b * NTOK + 1 + p) * DMODEL + cg] = val;
      }
}

// ---------------------------------------------------------------- attention
// Scores + softmax: P[bh, q, key] = softmax(Q Kt * 0.125), f16, keys padded 224
__global__ __launch_bounds__(256) void attn_scores_kernel(
    const float* __restrict__ qkv, _Float16* __restrict__ P) {
  __shared__ _Float16 Qs[16 * 64];
  __shared__ _Float16 Ks[NPAD * 64];
  __shared__ float Ss[16 * NPAD];
  const int bh = blockIdx.x, b = bh / NH_, h = bh % NH_;
  const int qt = blockIdx.y;
  const int tid = threadIdx.x, lane = tid & 31, wid = tid >> 5;
  const int r = lane & 15, kh = lane >> 4;

  for (int idx = tid; idx < NPAD * 16; idx += 256) {   // float4 over d
    int tok = idx >> 4, d4 = (idx & 15) * 4;
    float4 v = (tok < NTOK)
        ? *(const float4*)(qkv + (size_t)(b * NTOK + tok) * 2304 + DMODEL +
                           h * 64 + d4)
        : make_float4(0.f, 0.f, 0.f, 0.f);
    cvt_store4(&Ks[tok * 64 + d4], v);
  }
  for (int idx = tid; idx < 16 * 16; idx += 256) {
    int rr = idx >> 4, d4 = (idx & 15) * 4;
    int tok = qt * 16 + rr;
    float4 v = (tok < NTOK)
        ? *(const float4*)(qkv + (size_t)(b * NTOK + tok) * 2304 + h * 64 + d4)
        : make_float4(0.f, 0.f, 0.f, 0.f);
    cvt_store4(&Qs[rr * 64 + d4], v);
  }
  __syncthreads();

  for (int kt = wid; kt < NPAD / 16; kt += 8) {
    v8f s;
#pragma unroll
    for (int q = 0; q < 8; ++q) s[q] = 0.0f;
#pragma unroll
    for (int kk = 0; kk < 64; kk += 32) {
      v16h aq = frag_ld(&Qs[r * 64 + kk], kh);
      v16h bk = frag_ld(&Ks[(kt * 16 + r) * 64 + kk], kh);
      s = wmma32(aq, bk, s);
    }
#pragma unroll
    for (int v = 0; v < 8; ++v)
      Ss[(8 * kh + v) * NPAD + kt * 16 + r] = s[v] * 0.125f;
  }
  __syncthreads();

  {
    int row = wid * 2 + kh;
    float mx = -1e30f;
    for (int it = 0; it < NPAD / 16; ++it) {
      int cc = r + it * 16;
      if (cc < NTOK) mx = fmaxf(mx, Ss[row * NPAD + cc]);
    }
#pragma unroll
    for (int m = 1; m < 16; m <<= 1) mx = fmaxf(mx, __shfl_xor(mx, m, 32));
    float sum = 0.0f;
    for (int it = 0; it < NPAD / 16; ++it) {
      int cc = r + it * 16;
      if (cc < NTOK) sum += __expf(Ss[row * NPAD + cc] - mx);
    }
#pragma unroll
    for (int m = 1; m < 16; m <<= 1) sum += __shfl_xor(sum, m, 32);
    float inv = 1.0f / sum;
    size_t base = ((size_t)bh * PROWS + qt * 16 + row) * NPAD;
    for (int it = 0; it < NPAD / 16; ++it) {
      int cc = r + it * 16;
      float pv = (cc < NTOK) ? __expf(Ss[row * NPAD + cc] - mx) * inv : 0.0f;
      P[base + cc] = (_Float16)pv;
    }
  }
}

// O[b, q, h*64 + d] = P[16 x 224] @ V[224 x 64]
__global__ __launch_bounds__(256) void attn_pv_kernel(
    const float* __restrict__ qkv, const _Float16* __restrict__ P,
    float* __restrict__ o) {
  __shared__ _Float16 Vt[64 * NPAD];   // V transposed: row d, col key
  __shared__ _Float16 Ps[16 * NPAD];
  const int bh = blockIdx.x, b = bh / NH_, h = bh % NH_;
  const int qt = blockIdx.y;
  const int tid = threadIdx.x, lane = tid & 31, wid = tid >> 5;
  const int r = lane & 15, kh = lane >> 4;

  for (int idx = tid; idx < NPAD * 16; idx += 256) {
    int tok = idx >> 4, d4 = (idx & 15) * 4;
    float4 v = (tok < NTOK)
        ? *(const float4*)(qkv + (size_t)(b * NTOK + tok) * 2304 + 2 * DMODEL +
                           h * 64 + d4)
        : make_float4(0.f, 0.f, 0.f, 0.f);
    Vt[(d4 + 0) * NPAD + tok] = (_Float16)v.x;
    Vt[(d4 + 1) * NPAD + tok] = (_Float16)v.y;
    Vt[(d4 + 2) * NPAD + tok] = (_Float16)v.z;
    Vt[(d4 + 3) * NPAD + tok] = (_Float16)v.w;
  }
  size_t pbase = ((size_t)bh * PROWS + qt * 16) * NPAD;
  for (int i = tid; i < 16 * NPAD / 8; i += 256)        // v8h copies
    *(v8h*)&Ps[i * 8] = *(const v8h*)&P[pbase + i * 8];
  __syncthreads();

  if (wid < 4) {
    const int dt = wid;
    v8f acc;
#pragma unroll
    for (int q = 0; q < 8; ++q) acc[q] = 0.0f;
    for (int kt = 0; kt < NPAD / 32; ++kt) {
      v16h ap = frag_ld(&Ps[r * NPAD + kt * 32], kh);
      v16h bv = frag_ld(&Vt[(dt * 16 + r) * NPAD + kt * 32], kh);
      acc = wmma32(ap, bv, acc);
    }
#pragma unroll
    for (int v = 0; v < 8; ++v) {
      int tok = qt * 16 + 8 * kh + v;
      if (tok < NTOK)
        o[(size_t)(b * NTOK + tok) * DMODEL + h * 64 + dt * 16 + r] = acc[v];
    }
  }
}

// ---------------------------------------------------------------- launch
extern "C" void kernel_launch(void* const* d_in, const int* in_sizes, int n_in,
                              void* d_out, int out_size, void* d_ws,
                              size_t ws_size, hipStream_t stream) {
  (void)in_sizes; (void)n_in; (void)out_size; (void)ws_size;
  const float* x       = (const float*)d_in[0];
  const float* patch_w = (const float*)d_in[1];
  const float* patch_b = (const float*)d_in[2];
  const float* cls     = (const float*)d_in[3];
  const float* pos     = (const float*)d_in[4];
  const float* ln1_g   = (const float*)d_in[5];
  const float* ln1_b   = (const float*)d_in[6];
  const float* qkv_w   = (const float*)d_in[7];
  const float* qkv_b   = (const float*)d_in[8];
  const float* proj_w  = (const float*)d_in[9];
  const float* proj_b  = (const float*)d_in[10];
  const float* ln2_g   = (const float*)d_in[11];
  const float* ln2_b   = (const float*)d_in[12];
  const float* fc1_w   = (const float*)d_in[13];
  const float* fc1_b   = (const float*)d_in[14];
  const float* fc2_w   = (const float*)d_in[15];
  const float* fc2_b   = (const float*)d_in[16];
  const float* lnf_g   = (const float*)d_in[17];
  const float* lnf_b   = (const float*)d_in[18];
  float* out = (float*)d_out;

  char* ws = (char*)d_ws;
  size_t off = 0;
  auto alloc = [&](size_t bytes) -> void* {
    void* p = ws + off;
    off = (off + bytes + 255) & ~(size_t)255;
    return p;
  };
  float* t    = (float*)alloc((size_t)ROWS * DMODEL * 4);
  float* y    = (float*)alloc((size_t)ROWS * DMODEL * 4);
  float* qkvb = (float*)alloc((size_t)ROWS * 2304 * 4);
  float* obuf = (float*)alloc((size_t)ROWS * DMODEL * 4);
  float* hbuf = (float*)alloc((size_t)ROWS * HID_ * 4);
  _Float16* Pbuf = (_Float16*)alloc((size_t)BATCH * NH_ * PROWS * NPAD * 2);
  int* rmap = (int*)alloc((size_t)ROWS * 4);

  static const int hcnt[8]  = {5, 25, 30, 20, 10, 19, 34, 54};
  static const int heoff[8] = {0, 5, 30, 60, 80, 90, 109, 143};

  build_maps_kernel<<<1, 256, 0, stream>>>(rmap);
  cls_pos_kernel<<<BATCH, 256, 0, stream>>>(cls, pos, t);
  patch_embed_kernel<<<dim3(49, 6), 256, 0, stream>>>(x, patch_w, patch_b, pos, t);

  for (int l = 0; l < NLAYERS; ++l) {
    ln_kernel<<<ROWS, 256, 0, stream>>>(t, y, ln1_g + l * DMODEL,
                                        ln1_b + l * DMODEL);
    gemm_kernel<0, 0, 1><<<dim3(50, 9), 256, 0, stream>>>(
        y, qkv_w + (size_t)l * 3 * DMODEL * DMODEL,
        qkv_b + (size_t)l * 3 * DMODEL, qkvb, ROWS, 3 * DMODEL, DMODEL, nullptr);
    attn_scores_kernel<<<dim3(BATCH * NH_, 13), 256, 0, stream>>>(qkvb, Pbuf);
    attn_pv_kernel<<<dim3(BATCH * NH_, 13), 256, 0, stream>>>(qkvb, Pbuf, obuf);
    gemm_kernel<0, 0, 3><<<dim3(50, 3), 256, 0, stream>>>(
        obuf, proj_w + (size_t)l * DMODEL * DMODEL, proj_b + (size_t)l * DMODEL,
        t, ROWS, DMODEL, DMODEL, nullptr);
    ln_kernel<<<ROWS, 256, 0, stream>>>(t, y, ln2_g + l * DMODEL,
                                        ln2_b + l * DMODEL);
    for (int e = 0; e < NEXPERT; ++e) {
      int rows = 16 * hcnt[e];
      gemm_kernel<1, 1, 2><<<dim3((rows + 63) / 64, HID_ / 256), 256, 0, stream>>>(
          y, fc1_w + (size_t)(l * NEXPERT + e) * DMODEL * HID_,
          fc1_b + (size_t)(l * NEXPERT + e) * HID_,
          hbuf + (size_t)16 * heoff[e] * HID_, rows, HID_, DMODEL,
          rmap + 16 * heoff[e]);
    }
    for (int e = 0; e < NEXPERT; ++e) {
      int rows = 16 * hcnt[e];
      gemm_kernel<0, 1, 4><<<dim3((rows + 63) / 64, DMODEL / 256), 256, 0, stream>>>(
          hbuf + (size_t)16 * heoff[e] * HID_,
          fc2_w + (size_t)(l * NEXPERT + e) * HID_ * DMODEL,
          fc2_b + (size_t)(l * NEXPERT + e) * DMODEL, t, rows, DMODEL, HID_,
          rmap + 16 * heoff[e]);
    }
  }
  ln_kernel<<<ROWS, 256, 0, stream>>>(t, out, lnf_g, lnf_b);
}
